// AttentionHelper_704374636899
// MI455X (gfx1250) — matrix operand order repaired and checked by reading, not compiled, not run on hardware
//
#include <hip/hip_runtime.h>
#include <math.h>

// Problem constants (match reference)
#define BATCH 4
#define CC    512      // channels
#define LL    4096     // sequence length
#define SCALE 0.044194173824159216f   // 1/sqrt(512)

typedef __attribute__((ext_vector_type(16))) __bf16 v16bf;
typedef __attribute__((ext_vector_type(8)))  __bf16 bf16x8;
typedef __attribute__((ext_vector_type(4)))  __bf16 bf16x4;
typedef __attribute__((ext_vector_type(8)))  float  v8f;

static __device__ __forceinline__ v16bf concat8(bf16x8 lo, bf16x8 hi) {
    return __builtin_shufflevector(lo, hi, 0, 1, 2, 3, 4, 5, 6, 7,
                                            8, 9, 10, 11, 12, 13, 14, 15);
}

// ===========================================================================
// Preprocessing kernels (fast path): make k (=channel / =j) the contiguous
// axis in bf16 so WMMA fragments load as per-lane global_load_b128.
// ===========================================================================

// in: (C, L) fp32 per batch  ->  out: (L, C) bf16 per batch
__global__ void transpose_cvt_kernel(const float* __restrict__ in,
                                     __bf16* __restrict__ outp)
{
    __shared__ float tile[32][33];
    const int tx = threadIdx.x, ty = threadIdx.y;
    const size_t zb = (size_t)blockIdx.z * CC * LL;
    const int gx = blockIdx.x * 32;   // L index
    const int gy = blockIdx.y * 32;   // C index
#pragma unroll
    for (int r = ty; r < 32; r += 8)
        tile[r][tx] = in[zb + (size_t)(gy + r) * LL + gx + tx];
    __syncthreads();
#pragma unroll
    for (int r = ty; r < 32; r += 8)
        outp[zb + (size_t)(gx + r) * CC + gy + tx] = (__bf16)tile[tx][r];
}

// straight fp32 -> bf16 copy (layout preserved), 4 elements / thread
__global__ void cvt_bf16_kernel(const float* __restrict__ in,
                                __bf16* __restrict__ outp)
{
    const size_t idx = ((size_t)blockIdx.x * blockDim.x + threadIdx.x) * 4;
    const float4 v = *(const float4*)(in + idx);
    bf16x4 o = { (__bf16)v.x, (__bf16)v.y, (__bf16)v.z, (__bf16)v.w };
    *(bf16x4*)(outp + idx) = o;
}

// ===========================================================================
// Kernel 1 (fast): raw scores from transposed bf16 operands.
// S[b,i,j] = (sum_c Qt[i,c]*Kt[j,c]) * SCALE + log(mask[b,j]+1e-6)
// Wave tile: 32(i) x 64(j); all fragment feeds are global_load_b128.
// ===========================================================================
__global__ void scores_bf16_kernel(const __bf16* __restrict__ Qt,
                                   const __bf16* __restrict__ Kt,
                                   const float* __restrict__ mask,
                                   float* __restrict__ attn)
{
    const int lane = threadIdx.x & 31;
    const int wave = threadIdx.x >> 5;
    const int n    = lane & 15;
    const int h    = lane >> 4;
    const int b    = blockIdx.z;
    const int i0   = blockIdx.y * 32;
    const int j0   = (blockIdx.x * 8 + wave) * 64;

    const __bf16* __restrict__ Qb = Qt + (size_t)b * LL * CC;
    const __bf16* __restrict__ Kb = Kt + (size_t)b * LL * CC;

    v8f acc[2][4] = {};

    for (int c0 = 0; c0 < CC; c0 += 32) {
        // A fragments: A[m,k] = Qt[i0+16u+m, c0+k]; k = 16h + e
        v16bf afrag[2];
#pragma unroll
        for (int u = 0; u < 2; ++u) {
            const __bf16* ap = Qb + (size_t)(i0 + 16 * u + n) * CC + c0 + 16 * h;
            afrag[u] = concat8(*(const bf16x8*)ap, *(const bf16x8*)(ap + 8));
        }
#pragma unroll
        for (int t = 0; t < 4; ++t) {
            // B fragment: B[k,nn] = Kt[j0+16t+nn, c0+k]; k = 16h + e
            const __bf16* bp = Kb + (size_t)(j0 + 16 * t + n) * CC + c0 + 16 * h;
            const v16bf bfrag = concat8(*(const bf16x8*)bp, *(const bf16x8*)(bp + 8));
            acc[0][t] = __builtin_amdgcn_wmma_f32_16x16x32_bf16(
                false, afrag[0], false, bfrag, (short)0, acc[0][t], false, false);
            acc[1][t] = __builtin_amdgcn_wmma_f32_16x16x32_bf16(
                false, afrag[1], false, bfrag, (short)0, acc[1][t], false, false);
        }
    }

#pragma unroll
    for (int t = 0; t < 4; ++t) {
        const int j = j0 + 16 * t + n;
        const float lm = __logf(mask[(size_t)b * LL + j] + 1e-6f);
#pragma unroll
        for (int u = 0; u < 2; ++u)
#pragma unroll
            for (int r = 0; r < 8; ++r) {
                const int i = i0 + 16 * u + r + 8 * h;
                attn[((size_t)b * LL + i) * LL + j] = acc[u][t][r] * SCALE + lm;
            }
    }
}

// ===========================================================================
// Kernel 2: row softmax over j, then * mask[b,j]; in place. One 256-thread
// block per (b,i) row; values register-resident, LDS tree reductions.
// ===========================================================================
__global__ void attn_softmax_kernel(float* __restrict__ attn,
                                    const float* __restrict__ mask)
{
    const int row = blockIdx.x;           // 0 .. B*L-1
    const int b   = row / LL;
    float* __restrict__ rowp = attn + (size_t)row * LL;
    const float* __restrict__ mb = mask + (size_t)b * LL;

    const int t = threadIdx.x;
    float v[16];
    float mx = -INFINITY;
#pragma unroll
    for (int e = 0; e < 16; ++e) {
        v[e] = rowp[t * 16 + e];
        mx = fmaxf(mx, v[e]);
    }

    __shared__ float red[256];
    red[t] = mx;
    __syncthreads();
    for (int s = 128; s > 0; s >>= 1) {
        if (t < s) red[t] = fmaxf(red[t], red[t + s]);
        __syncthreads();
    }
    mx = red[0];
    __syncthreads();

    float sum = 0.f;
#pragma unroll
    for (int e = 0; e < 16; ++e) {
        v[e] = __expf(v[e] - mx);
        sum += v[e];
    }
    red[t] = sum;
    __syncthreads();
    for (int s = 128; s > 0; s >>= 1) {
        if (t < s) red[t] += red[t + s];
        __syncthreads();
    }
    const float inv = 1.0f / red[0];

#pragma unroll
    for (int e = 0; e < 16; ++e)
        rowp[t * 16 + e] = v[e] * inv * mb[t * 16 + e];
}

// ===========================================================================
// Kernel 3 (fast): out[b,c,i] = sum_j Vb[c,j] * A[i,j]
// Wave tile: 32(c) x 64(i).  A-feed: bf16 V, pure b128. B-feed: fp32
// attention rows, per-lane contiguous (b128 + pk-convert), with prefetch.
// ===========================================================================
__global__ void out_bf16_kernel(const __bf16* __restrict__ Vb,
                                const float* __restrict__ attn,
                                float* __restrict__ out)
{
    const int lane = threadIdx.x & 31;
    const int wave = threadIdx.x >> 5;
    const int n    = lane & 15;
    const int h    = lane >> 4;
    const int b    = blockIdx.z;
    const int c0   = blockIdx.y * 32;
    const int i0   = (blockIdx.x * 8 + wave) * 64;

    const __bf16* __restrict__ Vbb = Vb   + (size_t)b * CC * LL;
    const float* __restrict__  Ab  = attn + (size_t)b * LL * LL;

    v8f acc[2][4] = {};

    for (int j0 = 0; j0 < LL; j0 += 32) {
        // A fragments: A[m,k] = Vb[c0+16u+m, j0+k]; k = 16h + e
        v16bf afrag[2];
#pragma unroll
        for (int u = 0; u < 2; ++u) {
            const __bf16* ap = Vbb + (size_t)(c0 + 16 * u + n) * LL + j0 + 16 * h;
            afrag[u] = concat8(*(const bf16x8*)ap, *(const bf16x8*)(ap + 8));
        }
#pragma unroll
        for (int t = 0; t < 4; ++t) {
            // B fragment: B[k,nn] = A[i0+16t+nn, j0+k]; k = 16h+e -> 16
            // contiguous fp32 per lane
            const float* src = Ab + (size_t)(i0 + 16 * t + n) * LL + j0 + 16 * h;
            __builtin_prefetch(src + 32, 0, 3);
            v16bf bfrag;
#pragma unroll
            for (int e = 0; e < 16; ++e)
                bfrag[e] = (__bf16)src[e];
            acc[0][t] = __builtin_amdgcn_wmma_f32_16x16x32_bf16(
                false, afrag[0], false, bfrag, (short)0, acc[0][t], false, false);
            acc[1][t] = __builtin_amdgcn_wmma_f32_16x16x32_bf16(
                false, afrag[1], false, bfrag, (short)0, acc[1][t], false, false);
        }
    }

#pragma unroll
    for (int t = 0; t < 4; ++t)
#pragma unroll
        for (int u = 0; u < 2; ++u)
#pragma unroll
            for (int r = 0; r < 8; ++r) {
                const int c = c0 + 16 * u + r + 8 * h;
                out[((size_t)b * CC + c) * LL + i0 + 16 * t + n] = acc[u][t][r];
            }
}

// ===========================================================================
// Fallback kernels (small workspace): direct fp32->bf16 fragment builds.
// ===========================================================================
__global__ void scores_fp32_kernel(const float* __restrict__ Q,
                                   const float* __restrict__ K,
                                   const float* __restrict__ mask,
                                   float* __restrict__ attn)
{
    const int lane = threadIdx.x & 31;
    const int wave = threadIdx.x >> 5;
    const int n    = lane & 15;
    const int h    = lane >> 4;
    const int b    = blockIdx.z;
    const int i0   = blockIdx.y * 16;
    const int j0   = (blockIdx.x * 8 + wave) * 64;

    const float* __restrict__ Qb = Q + (size_t)b * CC * LL;
    const float* __restrict__ Kb = K + (size_t)b * CC * LL;

    v8f acc[4] = {};
    for (int c0 = 0; c0 < CC; c0 += 32) {
        v16bf afrag;
#pragma unroll
        for (int e = 0; e < 16; ++e) {
            const int k = 16 * h + e;
            afrag[e] = (__bf16)Qb[(size_t)(c0 + k) * LL + i0 + n];
        }
#pragma unroll
        for (int t = 0; t < 4; ++t) {
            v16bf bfrag;
#pragma unroll
            for (int e = 0; e < 16; ++e)
                bfrag[e] = (__bf16)Kb[(size_t)(c0 + 16 * h + e) * LL + j0 + 16 * t + n];
            acc[t] = __builtin_amdgcn_wmma_f32_16x16x32_bf16(
                false, afrag, false, bfrag, (short)0, acc[t], false, false);
        }
    }
#pragma unroll
    for (int t = 0; t < 4; ++t) {
        const int j = j0 + 16 * t + n;
        const float lm = __logf(mask[(size_t)b * LL + j] + 1e-6f);
#pragma unroll
        for (int r = 0; r < 8; ++r)
            attn[((size_t)b * LL + i0 + r + 8 * h) * LL + j] = acc[t][r] * SCALE + lm;
    }
}

__global__ void out_fp32_kernel(const float* __restrict__ V,
                                const float* __restrict__ attn,
                                float* __restrict__ out)
{
    const int lane = threadIdx.x & 31;
    const int wave = threadIdx.x >> 5;
    const int n    = lane & 15;
    const int h    = lane >> 4;
    const int b    = blockIdx.z;
    const int c0   = blockIdx.y * 16;
    const int i0   = (blockIdx.x * 8 + wave) * 64;

    const float* __restrict__ Vb = V    + (size_t)b * CC * LL;
    const float* __restrict__ Ab = attn + (size_t)b * LL * LL;

    v8f acc[4] = {};
    for (int j0 = 0; j0 < LL; j0 += 32) {
        v16bf afrag;
#pragma unroll
        for (int e = 0; e < 16; ++e) {
            const int k = 16 * h + e;
            afrag[e] = (__bf16)Vb[(size_t)(c0 + n) * LL + j0 + k];
        }
#pragma unroll
        for (int t = 0; t < 4; ++t) {
            const float* src = Ab + (size_t)(i0 + 16 * t + n) * LL + j0 + 16 * h;
            v16bf bfrag;
#pragma unroll
            for (int e = 0; e < 16; ++e)
                bfrag[e] = (__bf16)src[e];
            acc[t] = __builtin_amdgcn_wmma_f32_16x16x32_bf16(
                false, afrag, false, bfrag, (short)0, acc[t], false, false);
        }
    }
#pragma unroll
    for (int t = 0; t < 4; ++t)
#pragma unroll
        for (int r = 0; r < 8; ++r)
            out[((size_t)b * CC + c0 + r + 8 * h) * LL + i0 + 16 * t + n] = acc[t][r];
}

// ===========================================================================
extern "C" void kernel_launch(void* const* d_in, const int* in_sizes, int n_in,
                              void* d_out, int out_size, void* d_ws, size_t ws_size,
                              hipStream_t stream)
{
    (void)in_sizes; (void)n_in; (void)out_size;

    const float* Q    = (const float*)d_in[0];   // (B, C, L)
    const float* K    = (const float*)d_in[1];   // (B, C, L)
    const float* V    = (const float*)d_in[2];   // (B, C, L)
    const float* mask = (const float*)d_in[3];   // (B, 1, L)

    float* out  = (float*)d_out;                             // (B, C, L)
    float* attn = (float*)d_out + (size_t)BATCH * CC * LL;   // (B, L, L)

    const size_t elems = (size_t)BATCH * CC * LL;            // 8.4M
    const size_t need  = 3 * elems * sizeof(__bf16);         // ~48 MB

    if (ws_size >= need) {
        // ---- fast path: bf16 transposed operands in workspace ----
        __bf16* Qt = (__bf16*)d_ws;       // (B, L, C)
        __bf16* Kt = Qt + elems;          // (B, L, C)
        __bf16* Vb = Kt + elems;          // (B, C, L)

        dim3 tb(32, 8);
        dim3 tg(LL / 32, CC / 32, BATCH);
        transpose_cvt_kernel<<<tg, tb, 0, stream>>>(Q, Qt);
        transpose_cvt_kernel<<<tg, tb, 0, stream>>>(K, Kt);
        cvt_bf16_kernel<<<dim3(elems / (256 * 4)), 256, 0, stream>>>(V, Vb);

        dim3 g1(LL / 512, LL / 32, BATCH);
        scores_bf16_kernel<<<g1, 256, 0, stream>>>(Qt, Kt, mask, attn);

        attn_softmax_kernel<<<dim3(BATCH * LL), 256, 0, stream>>>(attn, mask);

        dim3 g3(LL / 512, CC / 32, BATCH);
        out_bf16_kernel<<<g3, 256, 0, stream>>>(Vb, attn, out);
    } else {
        // ---- fallback: direct fp32 fragment builds ----
        dim3 g1(LL / 512, LL / 16, BATCH);
        scores_fp32_kernel<<<g1, 256, 0, stream>>>(Q, K, mask, attn);

        attn_softmax_kernel<<<dim3(BATCH * LL), 256, 0, stream>>>(attn, mask);

        dim3 g3(LL / 512, CC / 16, BATCH);
        out_fp32_kernel<<<g3, 256, 0, stream>>>(V, attn, out);
    }
}
